// CapsNet_77318001262851
// MI455X (gfx1250) — compile-verified, matrix operand order checked by reading, not run on hardware
//
#include <hip/hip_runtime.h>
#include <cstdint>

typedef __attribute__((ext_vector_type(16))) _Float16 v16h;
typedef __attribute__((ext_vector_type(8)))  _Float16 v8h;
typedef __attribute__((ext_vector_type(8)))  float    v8f;
typedef __attribute__((ext_vector_type(4)))  unsigned int v4u;
typedef __attribute__((ext_vector_type(8)))  int      v8i;
typedef __attribute__((ext_vector_type(4)))  int      v4i;

// Problem constants
#define BATCH   512
#define C1      256        // conv1 out channels
#define S1      400        // 20*20
#define K2      20736      // 256*81
#define NCOL    18432      // 512*36
#define ROUTES  1152
#define CAPS    10
#define ODIM    16
#define IDIM    8

// ---------------------------------------------------------------------------
// Conv1: [512,1,28,28] * [256,1,9,9] -> relu -> f16 [512,256,20,20]
// ---------------------------------------------------------------------------
__global__ __launch_bounds__(256) void conv1_relu_f16(
    const float* __restrict__ data, const float* __restrict__ cw,
    const float* __restrict__ cb, _Float16* __restrict__ xh)
{
    __shared__ float din[784];
    __shared__ float wsh[81];
    const int oc = blockIdx.x, b = blockIdx.y, tid = threadIdx.x;
    for (int i = tid; i < 784; i += 256) din[i] = data[b * 784 + i];
    for (int i = tid; i < 81;  i += 256) wsh[i] = cw[oc * 81 + i];
    __syncthreads();
    const float bias = cb[oc];
    for (int s = tid; s < 400; s += 256) {
        const int oh = s / 20, ow = s % 20;
        float acc = bias;
#pragma unroll
        for (int kh = 0; kh < 9; ++kh)
#pragma unroll
            for (int kw = 0; kw < 9; ++kw)
                acc += din[(oh + kh) * 28 + ow + kw] * wsh[kh * 9 + kw];
        xh[(b * 256 + oc) * 400 + s] = (_Float16)fmaxf(acc, 0.0f);
    }
}

// ---------------------------------------------------------------------------
// pc_w f32 -> f16 ([256][20736] row-major = [OC][K])
// ---------------------------------------------------------------------------
__global__ __launch_bounds__(256) void cvt_w2_f16(
    const float* __restrict__ w, _Float16* __restrict__ wh)
{
    const int i = blockIdx.x * 256 + threadIdx.x;
    if (i < 256 * K2) wh[i] = (_Float16)w[i];
}

// ---------------------------------------------------------------------------
// PrimaryCaps conv as implicit GEMM with V_WMMA_F32_16X16X32_F16.
// C[oc,n] = sum_k w2h[oc,k] * im2col(xh)[k,n],  n = b*36 + oh*6 + ow.
// Block = 8 waves -> 16(OC) x 128(N); K chunks of 64, double-buffered LDS.
// A tile staged by the Tensor Data Mover (TDM), B tile gathered via registers.
// ---------------------------------------------------------------------------
#define KC    64           // K per chunk (2 WMMA steps)
#define LSTR  72           // padded LDS row stride in halves (64 data + 8 pad)
#define NREP  32           // B elements per thread per chunk: 128*64/256

__device__ __forceinline__ void tdm_load_A(
    const _Float16* gsrc, uint32_t lds_off)
{
    // D# group 0: count=1 | lds_addr | global_addr(57b) | type=2
    const uint64_t ga = (uint64_t)(uintptr_t)gsrc;
    v4u g0;
    g0[0] = 1u;                                   // count=1, user descriptor
    g0[1] = lds_off;                              // LDS byte address
    g0[2] = (uint32_t)ga;                         // global addr [31:0]
    g0[3] = (uint32_t)((ga >> 32) & 0x01FFFFFFu)  // global addr [56:32]
          | (2u << 30);                           // type = 2 (image)
    // D# group 1: data_size=2B, pad_enable, pad_interval=32 DW, pad_amount=4 DW
    // tensor_dim0 = K2, tensor_dim1 = 256, tile_dim0 = 64, tile_dim1 = 16,
    // tensor_dim0_stride = K2
    v8i g1;
    g1[0] = (int)((1u << 16) | (1u << 20) | (4u << 22) | (3u << 25));
    g1[1] = (int)((uint32_t)(K2 & 0xFFFF) << 16); // tensor_dim0[15:0]
    g1[2] = (int)((uint32_t)256 << 16);           // dim0[31:16]=0 | dim1[15:0]
    g1[3] = (int)((uint32_t)KC << 16);            // dim1[31:16]=0 | tile_dim0
    g1[4] = 16;                                   // tile_dim1=16, tile_dim2=0
    g1[5] = K2;                                   // tensor_dim0_stride[31:0]
    g1[6] = 0;                                    // stride hi | dim1_stride lo
    g1[7] = 0;
    v4i gz4 = {0, 0, 0, 0};                       // 2-D tensor: groups 2/3 unused
    v8i gz8 = {0, 0, 0, 0, 0, 0, 0, 0};
    __builtin_amdgcn_tensor_load_to_lds(g0, g1, gz4, gz4, gz8, 0);
}

__global__ __launch_bounds__(256) void conv2_wmma_f16(
    const _Float16* __restrict__ w2h, const _Float16* __restrict__ xh,
    const float* __restrict__ pcb, float* __restrict__ u)
{
    __shared__ alignas(16) _Float16 As[2][16 * LSTR];   // 4.5 KB
    __shared__ alignas(16) _Float16 Bs[2][128 * LSTR];  // 36 KB
    const int tid  = threadIdx.x;
    const int lane = tid & 31;
    const int wv   = tid >> 5;
    const int ocT  = blockIdx.y;          // 0..15
    const int n0   = blockIdx.x * 128;    // 144 tiles, exact
    const int m    = lane & 15;
    const int kb   = (lane >> 4) << 3;    // K-half base 0/8 (ISA striping)
    const int nl   = (wv << 4) + (lane & 15);

    // ---- hoisted im2col n-side addressing (divisions done once) ----
    const uint32_t kk    = (uint32_t)tid & 63u;   // k within chunk (per-thread)
    const uint32_t nrow0 = (uint32_t)tid >> 6;    // 0..3
    const uint32_t bso   = nrow0 * LSTR + kk;     // LDS store base offset
    uint32_t nbase[NREP];
#pragma unroll
    for (int rep = 0; rep < NREP; ++rep) {
        const uint32_t n  = (uint32_t)n0 + nrow0 + 4u * rep;
        const uint32_t bb = n / 36u, sp = n % 36u;
        const uint32_t oh = sp / 6u, ow = sp % 6u;
        nbase[rep] = bb * 102400u + oh * 40u + ow * 2u;
    }
    const _Float16* wrow = w2h + (size_t)(ocT * 16) * K2;

    auto kofs_of = [&](int k0) -> uint32_t {
        const uint32_t k  = (uint32_t)k0 + kk;
        const uint32_t ic = k / 81u, rem = k % 81u;
        const uint32_t kh = rem / 9u, kw = rem % 9u;
        return ic * 400u + kh * 20u + kw;
    };

    v8f acc = {};
    _Float16 btmp[NREP];
    const int nch = K2 / KC; // 324

    // ---- prologue: chunk 0 into buffer 0 ----
    {
        const uint32_t kofs = kofs_of(0);
#pragma unroll
        for (int rep = 0; rep < NREP; ++rep) btmp[rep] = xh[nbase[rep] + kofs];
#pragma unroll
        for (int rep = 0; rep < NREP; ++rep) Bs[0][bso + rep * (4 * LSTR)] = btmp[rep];
        if (wv == 0) {
            tdm_load_A(wrow, (uint32_t)(uintptr_t)&As[0][0]);
            __builtin_amdgcn_s_wait_tensorcnt(0);
        }
    }
    __syncthreads();

    for (int t = 0; t < nch; ++t) {
        const int cur = t & 1, nxt = cur ^ 1;
        const bool more = (t + 1) < nch;
        // issue next chunk's global gather early (hides under WMMA)
        if (more) {
            const uint32_t kofs = kofs_of((t + 1) * KC);
#pragma unroll
            for (int rep = 0; rep < NREP; ++rep) btmp[rep] = xh[nbase[rep] + kofs];
        }
        // compute: two 16x16x32 WMMA steps from current buffer
#pragma unroll
        for (int j = 0; j < 2; ++j) {
            const int ko = j * 32 + kb;
            const v8h alo = *reinterpret_cast<const v8h*>(&As[cur][m * LSTR + ko]);
            const v8h ahi = *reinterpret_cast<const v8h*>(&As[cur][m * LSTR + ko + 16]);
            const v8h blo = *reinterpret_cast<const v8h*>(&Bs[cur][nl * LSTR + ko]);
            const v8h bhi = *reinterpret_cast<const v8h*>(&Bs[cur][nl * LSTR + ko + 16]);
            v16h a, bm;
#pragma unroll
            for (int i = 0; i < 8; ++i) {
                a[i] = alo[i];  a[i + 8]  = ahi[i];
                bm[i] = blo[i]; bm[i + 8] = bhi[i];
            }
            acc = __builtin_amdgcn_wmma_f32_16x16x32_f16(
                false, a, false, bm, (short)0, acc, false, false);
        }
        // stage next chunk into the other buffer
        if (more) {
            if (wv == 0)
                tdm_load_A(wrow + (t + 1) * KC, (uint32_t)(uintptr_t)&As[nxt][0]);
#pragma unroll
            for (int rep = 0; rep < NREP; ++rep) Bs[nxt][bso + rep * (4 * LSTR)] = btmp[rep];
            if (wv == 0) __builtin_amdgcn_s_wait_tensorcnt(0);
        }
        __syncthreads();
    }

    // D layout: M = r + 8*(lane>=16), N = lane%16
    const int Nc  = n0 + (wv << 4) + (lane & 15);
    const int mhi = (lane >> 4) << 3;
    const uint32_t bb = (uint32_t)Nc / 36u, sp = (uint32_t)Nc % 36u;
#pragma unroll
    for (int r = 0; r < 8; ++r) {
        const int oc = ocT * 16 + mhi + r;
        u[bb * 9216u + (uint32_t)oc * 36u + sp] = acc[r] + pcb[oc];
    }
}

// ---------------------------------------------------------------------------
// squash over rows of 8 (in place), exactly as reference (no eps)
// ---------------------------------------------------------------------------
__global__ __launch_bounds__(256) void squash_rows8(float* __restrict__ u)
{
    const int r = blockIdx.x * 256 + threadIdx.x;
    if (r >= BATCH * ROUTES) return;
    float* p = u + (size_t)r * 8;
    float sn = 0.0f;
#pragma unroll
    for (int i = 0; i < 8; ++i) sn += p[i] * p[i];
    const float sc = sn / ((1.0f + sn) * sqrtf(sn));
#pragma unroll
    for (int i = 0; i < 8; ++i) p[i] *= sc;
}

__global__ __launch_bounds__(256) void zero_bij(float* __restrict__ bij)
{
    const int i = blockIdx.x * 256 + threadIdx.x;
    if (i < ROUTES * CAPS) bij[i] = 0.0f;
}

// softmax over routes (axis 0); one block per capsule column
__global__ __launch_bounds__(256) void softmax_routes(
    const float* __restrict__ bij, float* __restrict__ cij)
{
    __shared__ float red[256];
    const int c = blockIdx.x, t = threadIdx.x;
    float mx = -3.402823466e+38f;
    for (int r = t; r < ROUTES; r += 256) mx = fmaxf(mx, bij[r * CAPS + c]);
    red[t] = mx; __syncthreads();
    for (int s = 128; s > 0; s >>= 1) { if (t < s) red[t] = fmaxf(red[t], red[t + s]); __syncthreads(); }
    mx = red[0]; __syncthreads();
    float sum = 0.0f;
    for (int r = t; r < ROUTES; r += 256) sum += expf(bij[r * CAPS + c] - mx);
    red[t] = sum; __syncthreads();
    for (int s = 128; s > 0; s >>= 1) { if (t < s) red[t] += red[t + s]; __syncthreads(); }
    const float Z = red[0];
    for (int r = t; r < ROUTES; r += 256)
        cij[r * CAPS + c] = expf(bij[r * CAPS + c] - mx) / Z;
}

// s[b,c,o] = sum_r c[r,c] * (W[r,c,o,:] . u[b,r,:]);  v = s|s|/(1+s^2)
__global__ __launch_bounds__(256) void route_s_v(
    const float* __restrict__ W, const float* __restrict__ u,
    const float* __restrict__ cij, float* __restrict__ v)
{
    const int idx = blockIdx.x * 256 + threadIdx.x;
    if (idx >= BATCH * CAPS * ODIM) return;
    const int b = idx / 160, co = idx % 160;
    const int c = co >> 4, o = co & 15;
    const float* urow = u + (size_t)b * 9216;
    float acc = 0.0f;
    for (int r = 0; r < ROUTES; ++r) {
        const float* wp = W + ((size_t)(r * CAPS + c)) * 128 + o * 8;
        const float* uu = urow + r * 8;
        float uh = 0.0f;
#pragma unroll
        for (int i = 0; i < 8; ++i) uh += wp[i] * uu[i];
        acc += cij[r * CAPS + c] * uh;
    }
    const float s2 = acc * acc;
    v[idx] = s2 * acc / ((1.0f + s2) * sqrtf(s2));
}

// b_ij[r,c] += mean_b ( (W[r,c]@u[b,r]) . v[b,c] ), u_hat recomputed on the fly
__global__ __launch_bounds__(256) void route_update_b(
    const float* __restrict__ W, const float* __restrict__ u,
    const float* __restrict__ v, float* __restrict__ bij)
{
    const int idx = blockIdx.x * 256 + threadIdx.x;
    if (idx >= ROUTES * CAPS) return;
    const int r = idx / CAPS, c = idx % CAPS;
    const float* wp = W + ((size_t)(r * CAPS + c)) * 128;
    float Wl[128];
#pragma unroll
    for (int j = 0; j < 128; ++j) Wl[j] = wp[j];
    float acc = 0.0f;
    for (int b = 0; b < BATCH; ++b) {
        const float* uu = u + (size_t)b * 9216 + r * 8;
        float u8[8];
#pragma unroll
        for (int i = 0; i < 8; ++i) u8[i] = uu[i];
        const float* vb = v + (size_t)b * 160 + c * 16;
#pragma unroll
        for (int o = 0; o < 16; ++o) {
            float uh = 0.0f;
#pragma unroll
            for (int i = 0; i < 8; ++i) uh += Wl[o * 8 + i] * u8[i];
            acc += uh * vb[o];
        }
    }
    bij[idx] += acc * (1.0f / 512.0f);
}

// Encoder: h = relu(v @ w1^T + b1) ; out = sigmoid(h @ w2^T + b2)
__global__ __launch_bounds__(256) void enc_fc1(
    const float* __restrict__ v, const float* __restrict__ w1,
    const float* __restrict__ b1, float* __restrict__ h)
{
    const int idx = blockIdx.x * 256 + threadIdx.x;
    if (idx >= BATCH * 512) return;
    const int b = idx >> 9, j = idx & 511;
    const float* vp = v + (size_t)b * 160;
    const float* wp = w1 + (size_t)j * 160;
    float acc = b1[j];
    for (int k = 0; k < 160; ++k) acc += vp[k] * wp[k];
    h[idx] = fmaxf(acc, 0.0f);
}

__global__ __launch_bounds__(256) void enc_fc2(
    const float* __restrict__ h, const float* __restrict__ w2,
    const float* __restrict__ b2, float* __restrict__ out)
{
    const int idx = blockIdx.x * 256 + threadIdx.x;
    if (idx >= BATCH * 9) return;
    const int b = idx / 9, j = idx % 9;
    const float* hp = h + (size_t)b * 512;
    const float* wp = w2 + (size_t)j * 512;
    float acc = b2[j];
    for (int k = 0; k < 512; ++k) acc += hp[k] * wp[k];
    out[idx] = 1.0f / (1.0f + expf(-acc));
}

// ---------------------------------------------------------------------------
extern "C" void kernel_launch(void* const* d_in, const int* in_sizes, int n_in,
                              void* d_out, int out_size, void* d_ws, size_t ws_size,
                              hipStream_t stream)
{
    (void)in_sizes; (void)n_in; (void)out_size; (void)ws_size;
    const float* data   = (const float*)d_in[0];
    const float* conv_w = (const float*)d_in[1];
    const float* conv_b = (const float*)d_in[2];
    const float* pc_w   = (const float*)d_in[3];
    const float* pc_b   = (const float*)d_in[4];
    const float* W      = (const float*)d_in[5];
    const float* enc_w1 = (const float*)d_in[6];
    const float* enc_b1 = (const float*)d_in[7];
    const float* enc_w2 = (const float*)d_in[8];
    const float* enc_b2 = (const float*)d_in[9];
    float* out = (float*)d_out;

    // Workspace carve-out (aligned to 256B)
    char* p = (char*)d_ws;
    auto take = [&](size_t bytes) { char* q = p; p += (bytes + 255) & ~(size_t)255; return q; };
    _Float16* xh  = (_Float16*)take((size_t)BATCH * C1 * S1 * 2);   // 104.9 MB
    _Float16* w2h = (_Float16*)take((size_t)C1 * K2 * 2);           //  10.6 MB
    float*    u   = (float*)take((size_t)BATCH * 9216 * 4);         //  18.9 MB
    float*    bij = (float*)take((size_t)ROUTES * CAPS * 4);
    float*    cij = (float*)take((size_t)ROUTES * CAPS * 4);
    float*    v   = (float*)take((size_t)BATCH * 160 * 4);
    float*    h   = (float*)take((size_t)BATCH * 512 * 4);

    conv1_relu_f16<<<dim3(C1, BATCH), 256, 0, stream>>>(data, conv_w, conv_b, xh);
    cvt_w2_f16<<<(C1 * K2) / 256, 256, 0, stream>>>(pc_w, w2h);
    conv2_wmma_f16<<<dim3(NCOL / 128, C1 / 16), 256, 0, stream>>>(w2h, xh, pc_b, u);
    squash_rows8<<<(BATCH * ROUTES) / 256, 256, 0, stream>>>(u);
    zero_bij<<<(ROUTES * CAPS + 255) / 256, 256, 0, stream>>>(bij);
    for (int it = 0; it < 3; ++it) {
        softmax_routes<<<CAPS, 256, 0, stream>>>(bij, cij);
        route_s_v<<<(BATCH * CAPS * ODIM) / 256, 256, 0, stream>>>(W, u, cij, v);
        if (it < 2)
            route_update_b<<<(ROUTES * CAPS + 255) / 256, 256, 0, stream>>>(W, u, v, bij);
    }
    enc_fc1<<<(BATCH * 512) / 256, 256, 0, stream>>>(v, enc_w1, enc_b1, h);
    enc_fc2<<<(BATCH * 9 + 255) / 256, 256, 0, stream>>>(h, enc_w2, enc_b2, out);
}
